// DialogueRNNCell_test_38285338476834
// MI455X (gfx1250) — matrix operand order repaired and checked by reading, not compiled, not run on hardware
//
#include <hip/hip_runtime.h>
#include <hip/hip_bf16.h>
#include <cstddef>

// ---------------------------------------------------------------------------
// DialogueRNN cell for MI455X (gfx1250, wave32).
//   - 7 GRU/projection GEMMs on v_wmma_f32_16x16x32_bf16, fp32 accumulate.
//     Operands pre-converted to bf16 once per launch; fragments loaded as
//     contiguous b128s in one contiguous K order for A and B; each wave computes a
//     16x64 strip (A fragment reused across 4 WMMAs).
//   - Attention passes are HBM-bound streamers (g_hist 200MB + q_hist 400MB,
//     each read twice -> ~1.2GB -> ~52us at 23.3 TB/s). Alpha vectors staged
//     into LDS with global_load_async_to_lds_b32 (ASYNCcnt).
// ---------------------------------------------------------------------------

typedef __attribute__((ext_vector_type(16))) __bf16 v16bf;
typedef __attribute__((ext_vector_type(8)))  float  v8f;

#define BB    512
#define TT    200
#define DM    1024
#define DG    512
#define DP    512
#define DE    512
#define K_CAT 1536   // DM + DP == DM + DG
#define M3    1536   // 3 * D

union Frag { v16bf v; uint4 q[2]; };

// ---------------------------------------------------------------------------
// GEMM: Out[N x M](f32) = Xb[N x K](bf16) @ Wb[M x K](bf16)^T + bias[M]
// One wave per 16(N) x 64(M) strip: 1 A fragment, 4 B fragments, 4 WMMAs
// per 32-wide K step. Requires N%16==0, M%64==0, K%32==0 (all call sites ok).
// ---------------------------------------------------------------------------
__global__ __launch_bounds__(256) void gemm_bias_wmma(
    const __bf16* __restrict__ Xb, const __bf16* __restrict__ Wb,
    const float* __restrict__ bias, float* __restrict__ Out,
    int N, int K, int M)
{
    const int wave = threadIdx.x >> 5;          // 0..7
    const int lane = threadIdx.x & 31;
    const int groupsM = M >> 6;                 // 64-wide column groups
    const int strips  = (N >> 4) * groupsM;
    const int strip   = blockIdx.x * 8 + wave;  // wave-uniform
    if (strip >= strips) return;                // uniform exit: EXEC all-1s at WMMA
    const int gm = strip % groupsM;             // column group
    const int tm = strip / groupsM;             // 16-row tile
    const int l16  = lane & 15;
    const int half = lane >> 4;
    const int ha = half << 4;                   // A: K base offset per half
    const int hb = half << 4;                   // B: K base offset per half

    const __bf16* __restrict__ xrow = Xb + (size_t)((tm << 4) + l16) * K;
    const __bf16* __restrict__ w0   = Wb + (size_t)((gm << 6) + l16) * K;
    const size_t wstep = (size_t)16 * K;        // next 16-column tile of W

    v8f acc0 = {}, acc1 = {}, acc2 = {}, acc3 = {};
    for (int k0 = 0; k0 < K; k0 += 32) {
        __builtin_prefetch(xrow + k0 + 32, 0, 1);    // global_prefetch_b8
        Frag a, b0, b1, b2, b3;
        // A (16x32 bf16): lane -> row M=l%16; elems 0..15 = K ha..ha+15
        // (same contiguous K order as B)
        a.q[0] = *(const uint4*)(xrow + k0 + ha);
        a.q[1] = *(const uint4*)(xrow + k0 + ha + 8);
        // B (32x16 bf16): lane -> col N=l%16; elems 0..15 = K hb..hb+15
        const __bf16* wk = w0 + k0 + hb;
        b0.q[0] = *(const uint4*)(wk);
        b0.q[1] = *(const uint4*)(wk + 8);
        b1.q[0] = *(const uint4*)(wk + wstep);
        b1.q[1] = *(const uint4*)(wk + wstep + 8);
        b2.q[0] = *(const uint4*)(wk + 2 * wstep);
        b2.q[1] = *(const uint4*)(wk + 2 * wstep + 8);
        b3.q[0] = *(const uint4*)(wk + 3 * wstep);
        b3.q[1] = *(const uint4*)(wk + 3 * wstep + 8);
        acc0 = __builtin_amdgcn_wmma_f32_16x16x32_bf16(false, a.v, false, b0.v, (short)0, acc0, false, false);
        acc1 = __builtin_amdgcn_wmma_f32_16x16x32_bf16(false, a.v, false, b1.v, (short)0, acc1, false, false);
        acc2 = __builtin_amdgcn_wmma_f32_16x16x32_bf16(false, a.v, false, b2.v, (short)0, acc2, false, false);
        acc3 = __builtin_amdgcn_wmma_f32_16x16x32_bf16(false, a.v, false, b3.v, (short)0, acc3, false, false);
    }

    // D layout: lane -> N=l%16 ; VGPR v -> M = v + 8*(lane/16)
    const int mrow = (tm << 4) + (half << 3);
    v8f accs[4] = { acc0, acc1, acc2, acc3 };
#pragma unroll
    for (int j = 0; j < 4; ++j) {
        const int n = (gm << 6) + (j << 4) + l16;
        const float bv = bias ? bias[n] : 0.0f;
#pragma unroll
        for (int v = 0; v < 8; ++v)
            Out[(size_t)(mrow + v) * M + n] = accs[j][v] + bv;
    }
}

// ---------------------------------------------------------------------------
// fp32 -> bf16 bulk conversion (4 elems/thread; n % 1024 == 0)
// ---------------------------------------------------------------------------
__global__ __launch_bounds__(256) void f32_to_bf16_kernel(
    const float* __restrict__ src, __bf16* __restrict__ dst, long n)
{
    const long i = ((long)blockIdx.x * 256 + threadIdx.x) * 4;
    if (i >= n) return;
    const float4 v = *(const float4*)(src + i);
    dst[i]     = (__bf16)v.x;
    dst[i + 1] = (__bf16)v.y;
    dst[i + 2] = (__bf16)v.z;
    dst[i + 3] = (__bf16)v.w;
}

// ---------------------------------------------------------------------------
// out[t*R + r] = dot(vec_r, Y[t, r, :])   (vec_stride==0 -> broadcast vector)
// ---------------------------------------------------------------------------
__global__ __launch_bounds__(256) void rowdot_kernel(
    const float* __restrict__ vecs, int vec_stride,
    const float* __restrict__ Y, float* __restrict__ out,
    int T, int R, int D)
{
    const int wave = threadIdx.x >> 5;
    const int lane = threadIdx.x & 31;
    const long idx = (long)blockIdx.x * 8 + wave;
    if (idx >= (long)T * R) return;
    const int r = (int)(idx % R);
    const float* __restrict__ v = vecs + (size_t)r * vec_stride;
    const float* __restrict__ y = Y + (size_t)idx * D;
    float s = 0.f;
    for (int d = lane; d < D; d += 32) s += v[d] * y[d];
#pragma unroll
    for (int off = 16; off > 0; off >>= 1) s += __shfl_down(s, off, 32);
    if (lane == 0) out[idx] = s;
}

// ---------------------------------------------------------------------------
// Column softmax over T (in-place on s[T x R]); optional transposed copy.
// ---------------------------------------------------------------------------
__global__ __launch_bounds__(256) void softmax_t_kernel(
    float* __restrict__ s, float* __restrict__ alpha_out, int T, int R)
{
    const int r = blockIdx.x;
    const int tid = threadIdx.x;
    __shared__ float red[256];

    float m = -3.0e38f;
    for (int t = tid; t < T; t += 256) m = fmaxf(m, s[(size_t)t * R + r]);
    red[tid] = m; __syncthreads();
    for (int w = 128; w > 0; w >>= 1) {
        if (tid < w) red[tid] = fmaxf(red[tid], red[tid + w]);
        __syncthreads();
    }
    m = red[0]; __syncthreads();

    float sum = 0.f;
    for (int t = tid; t < T; t += 256) sum += __expf(s[(size_t)t * R + r] - m);
    red[tid] = sum; __syncthreads();
    for (int w = 128; w > 0; w >>= 1) {
        if (tid < w) red[tid] += red[tid + w];
        __syncthreads();
    }
    const float inv = 1.0f / red[0];

    for (int t = tid; t < T; t += 256) {
        const float a = __expf(s[(size_t)t * R + r] - m) * inv;
        s[(size_t)t * R + r] = a;
        if (alpha_out) alpha_out[(size_t)r * T + t] = a;
    }
}

// ---------------------------------------------------------------------------
// out[r, :] = sum_t alpha[t, r] * Y[t, r, :]
// Alpha column staged into LDS with gfx1250 async-to-LDS copy (T <= 256).
// ---------------------------------------------------------------------------
__global__ __launch_bounds__(256) void weighted_sum_kernel(
    const float* __restrict__ alpha, const float* __restrict__ Y,
    float* __restrict__ out, int T, int R, int D)
{
    const int r = blockIdx.x;
    __shared__ float sa[256];
    if ((int)threadIdx.x < T) {
        // per-lane: LDS[sa + tid*4] <- alpha[tid*R + r]   (ASYNCcnt-tracked)
        unsigned lds_off = (unsigned)(size_t)&sa[threadIdx.x];
        unsigned long long ga =
            (unsigned long long)(const void*)(alpha + (size_t)threadIdx.x * R + r);
        asm volatile("global_load_async_to_lds_b32 %0, %1, off"
                     :: "v"(lds_off), "v"(ga) : "memory");
    }
    asm volatile("s_wait_asynccnt 0x0" ::: "memory");
    __syncthreads();

    for (int d = threadIdx.x; d < D; d += 256) {
        float acc = 0.f;
        for (int t = 0; t < T; ++t)
            acc += sa[t] * Y[((size_t)t * R + r) * D + d];
        out[(size_t)r * D + d] = acc;
    }
}

// ---------------------------------------------------------------------------
// GRU gate math:  r=sig(ir+hr); z=sig(iz+hz); n=tanh(in+r*hn)
// h' = (1-z)*n + z*h ; optional per-row mask mix: out = h*(1-m) + h'*m
// ---------------------------------------------------------------------------
__global__ __launch_bounds__(256) void gru_pointwise_kernel(
    const float* __restrict__ gi, const float* __restrict__ gh,
    const float* __restrict__ h, const float* __restrict__ mask,
    float* __restrict__ out, int N, int D)
{
    const long i = (long)blockIdx.x * blockDim.x + threadIdx.x;
    if (i >= (long)N * D) return;
    const int n = (int)(i / D);
    const int d = (int)(i % D);
    const size_t base = (size_t)n * 3 * D;
    const float ir = gi[base + d],          hr = gh[base + d];
    const float iz = gi[base + D + d],      hz = gh[base + D + d];
    const float in_ = gi[base + 2 * D + d], hn = gh[base + 2 * D + d];
    const float rg = 1.f / (1.f + __expf(-(ir + hr)));
    const float zg = 1.f / (1.f + __expf(-(iz + hz)));
    const float ng = tanhf(in_ + rg * hn);
    const float hv = h[(size_t)n * D + d];
    float hnew = (1.f - zg) * ng + zg * hv;
    if (mask) { const float mv = mask[n]; hnew = hv * (1.f - mv) + hnew * mv; }
    out[i] = hnew;
}

// ---------------------------------------------------------------------------
// Glue kernels (emit bf16 GEMM activations directly)
// ---------------------------------------------------------------------------
__global__ __launch_bounds__(256) void build_x1_kernel(
    const float* __restrict__ U, const float* __restrict__ qmask,
    const float* __restrict__ q0, __bf16* __restrict__ X1)
{
    const int b = blockIdx.x;                       // 0..B-1
    const int sel = (qmask[b * 2 + 1] > qmask[b * 2 + 0]) ? 1 : 0;  // argmax (one-hot)
    const float* __restrict__ qsel = q0 + ((size_t)b * 2 + sel) * DP;
    __bf16* __restrict__ xr = X1 + (size_t)b * K_CAT;
    for (int i = threadIdx.x; i < DM; i += 256) xr[i] = (__bf16)U[(size_t)b * DM + i];
    for (int i = threadIdx.x; i < DP; i += 256) xr[DM + i] = (__bf16)qsel[i];
}

__global__ __launch_bounds__(256) void build_x2_kernel(
    const float* __restrict__ U, const float* __restrict__ c_,
    __bf16* __restrict__ X2)
{
    const int r = blockIdx.x;                       // 0..B*2-1
    const int b = r >> 1;
    __bf16* __restrict__ xr = X2 + (size_t)r * K_CAT;
    for (int i = threadIdx.x; i < DM; i += 256) xr[i] = (__bf16)U[(size_t)b * DM + i];
    for (int i = threadIdx.x; i < DG; i += 256) xr[DM + i] = (__bf16)c_[(size_t)b * DG + i];
}

__global__ __launch_bounds__(256) void swap_party_kernel(
    const float* __restrict__ e0, __bf16* __restrict__ e0sw)
{
    const int r = blockIdx.x;                       // 0..B*2-1
    const int rs = r ^ 1;                           // e0[:, ::-1, :]
    for (int i = threadIdx.x; i < DE; i += 256)
        e0sw[(size_t)r * DE + i] = (__bf16)e0[(size_t)rs * DE + i];
}

// ---------------------------------------------------------------------------
// Host launcher
// ---------------------------------------------------------------------------
extern "C" void kernel_launch(void* const* d_in, const int* in_sizes, int n_in,
                              void* d_out, int out_size, void* d_ws, size_t ws_size,
                              hipStream_t stream)
{
    const float* U       = (const float*)d_in[0];
    const float* qmask   = (const float*)d_in[1];
    const float* g_hist  = (const float*)d_in[2];
    const float* q0      = (const float*)d_in[3];
    const float* q_hist  = (const float*)d_in[4];
    const float* e0      = (const float*)d_in[5];
    const float* w_ih_g  = (const float*)d_in[6];
    const float* w_hh_g  = (const float*)d_in[7];
    const float* b_ih_g  = (const float*)d_in[8];
    const float* b_hh_g  = (const float*)d_in[9];
    const float* w_ih_p  = (const float*)d_in[10];
    const float* w_hh_p  = (const float*)d_in[11];
    const float* b_ih_p  = (const float*)d_in[12];
    const float* b_hh_p  = (const float*)d_in[13];
    const float* w_ih_e  = (const float*)d_in[14];
    const float* w_hh_e  = (const float*)d_in[15];
    const float* b_ih_e  = (const float*)d_in[16];
    const float* b_hh_e  = (const float*)d_in[17];
    const float* w_att   = (const float*)d_in[18];
    const float* w_trans = (const float*)d_in[19];
    (void)in_sizes; (void)n_in; (void)out_size; (void)ws_size;

    // d_out: g_(B*DG) | q_(B*2*DP) | e_(B*2*DE) | c_(B*DG) | alpha(B*1*T)
    float* outG     = (float*)d_out;
    float* outQ     = outG + (size_t)BB * DG;
    float* outE     = outQ + (size_t)BB * 2 * DP;
    float* outC     = outE + (size_t)BB * 2 * DE;
    float* outAlpha = outC + (size_t)BB * DG;

    // ---- workspace: fp32 region then bf16 region (~42 MB total) ----
    float* wsf = (float*)d_ws;
    size_t f = 0;
    float* bufGI  = wsf + f; f += (size_t)1024 * M3;
    float* bufGH  = wsf + f; f += (size_t)1024 * M3;
    float* alphaG = wsf + f; f += (size_t)TT * BB;
    float* alphaP = wsf + f; f += (size_t)TT * BB * 2;
    float* xt     = wsf + f; f += (size_t)1024 * DP;
    float* Qf     = wsf + f; f += (size_t)1024 * DP;
    __bf16* wsb = (__bf16*)(wsf + f);
    size_t g = 0;
    __bf16* Xb    = wsb + g; g += (size_t)1024 * K_CAT;   // X1 then X2
    __bf16* Wihg  = wsb + g; g += (size_t)M3 * K_CAT;
    __bf16* Whhg  = wsb + g; g += (size_t)M3 * DG;
    __bf16* Wihp  = wsb + g; g += (size_t)M3 * K_CAT;
    __bf16* Whhp  = wsb + g; g += (size_t)M3 * DP;
    __bf16* Wihe  = wsb + g; g += (size_t)M3 * DP;
    __bf16* Whhe  = wsb + g; g += (size_t)M3 * DE;
    __bf16* Wtr   = wsb + g; g += (size_t)DP * DE;
    __bf16* hgb   = wsb + g; g += (size_t)BB * DG;        // g_hist[-1]
    __bf16* q0b   = wsb + g; g += (size_t)1024 * DP;
    __bf16* e0b   = wsb + g; g += (size_t)1024 * DE;
    __bf16* e0swb = wsb + g; g += (size_t)1024 * DE;
    __bf16* Qb    = wsb + g; g += (size_t)1024 * DP;
    (void)g;

    const float* g_last = g_hist + (size_t)(TT - 1) * BB * DG;

    auto cvt = [&](const float* s, __bf16* d, size_t n) {
        f32_to_bf16_kernel<<<(unsigned)(n / 1024), 256, 0, stream>>>(s, d, (long)n);
    };

    // ---- one-time bf16 conversions (weights + hidden states) ----
    cvt(w_ih_g, Wihg, (size_t)M3 * K_CAT);
    cvt(w_hh_g, Whhg, (size_t)M3 * DG);
    cvt(w_ih_p, Wihp, (size_t)M3 * K_CAT);
    cvt(w_hh_p, Whhp, (size_t)M3 * DP);
    cvt(w_ih_e, Wihe, (size_t)M3 * DP);
    cvt(w_hh_e, Whhe, (size_t)M3 * DE);
    cvt(w_trans, Wtr, (size_t)DP * DE);
    cvt(g_last, hgb, (size_t)BB * DG);
    cvt(q0, q0b, (size_t)1024 * DP);
    cvt(e0, e0b, (size_t)1024 * DE);

    // ---- global GRU: g_ = GRU([U, q0_sel], g_hist[-1]) ----
    build_x1_kernel<<<BB, 256, 0, stream>>>(U, qmask, q0, Xb);
    gemm_bias_wmma<<<(32 * 24) / 8, 256, 0, stream>>>(Xb, Wihg, b_ih_g, bufGI, BB, K_CAT, M3);
    gemm_bias_wmma<<<(32 * 24) / 8, 256, 0, stream>>>(hgb, Whhg, b_hh_g, bufGH, BB, DG, M3);
    gru_pointwise_kernel<<<(BB * DG) / 256, 256, 0, stream>>>(bufGI, bufGH, g_last, nullptr, outG, BB, DG);

    // ---- attention over g_hist: alpha, c_ ----
    rowdot_kernel<<<(TT * BB) / 8, 256, 0, stream>>>(w_att, 0, g_hist, alphaG, TT, BB, DG);
    softmax_t_kernel<<<BB, 256, 0, stream>>>(alphaG, outAlpha, TT, BB);
    weighted_sum_kernel<<<BB, 256, 0, stream>>>(alphaG, g_hist, outC, TT, BB, DG);

    // ---- party GRU: qs_ = GRU([U, c_] rep, q0), masked mix -> q_ ----
    build_x2_kernel<<<BB * 2, 256, 0, stream>>>(U, outC, Xb);
    gemm_bias_wmma<<<(64 * 24) / 8, 256, 0, stream>>>(Xb, Wihp, b_ih_p, bufGI, BB * 2, K_CAT, M3);
    gemm_bias_wmma<<<(64 * 24) / 8, 256, 0, stream>>>(q0b, Whhp, b_hh_p, bufGH, BB * 2, DP, M3);
    gru_pointwise_kernel<<<(BB * 2 * DP) / 256, 256, 0, stream>>>(bufGI, bufGH, q0, qmask, outQ, BB * 2, DP);

    // ---- emotion attention over q_hist ----
    swap_party_kernel<<<BB * 2, 256, 0, stream>>>(e0, e0swb);
    gemm_bias_wmma<<<(64 * 8) / 8, 256, 0, stream>>>(e0swb, Wtr, nullptr, xt, BB * 2, DE, DP);
    rowdot_kernel<<<(TT * BB * 2) / 8, 256, 0, stream>>>(xt, DP, q_hist, alphaP, TT, BB * 2, DP);
    softmax_t_kernel<<<BB * 2, 256, 0, stream>>>(alphaP, nullptr, TT, BB * 2);
    weighted_sum_kernel<<<BB * 2, 256, 0, stream>>>(alphaP, q_hist, Qf, TT, BB * 2, DP);

    // ---- emotion GRU: e_ = GRU(Q, e0) ----
    cvt(Qf, Qb, (size_t)1024 * DP);
    gemm_bias_wmma<<<(64 * 24) / 8, 256, 0, stream>>>(Qb, Wihe, b_ih_e, bufGI, BB * 2, DP, M3);
    gemm_bias_wmma<<<(64 * 24) / 8, 256, 0, stream>>>(e0b, Whhe, b_hh_e, bufGH, BB * 2, DE, M3);
    gru_pointwise_kernel<<<(BB * 2 * DE) / 256, 256, 0, stream>>>(bufGI, bufGH, e0, nullptr, outE, BB * 2, DE);
}
